// Head_54631984005784
// MI455X (gfx1250) — compile-verified, matrix operand order checked by reading, not run on hardware
//
#include <hip/hip_runtime.h>
#include <hip/hip_bf16.h>

// ---------------------------------------------------------------------------
// Causal single-head attention for MI455X (gfx1250, wave32, WMMA bf16).
//   x:[8,2048,1024] f32, Wk/Wq/Wv:[1024,128] f32  ->  out:[8,2048,128] f32
// Kernel 1: QKV projection (bf16 WMMA GEMM), writes Q,K row-major bf16 and
//           V transposed ([H][T]) bf16 into workspace.
// Kernel 2: flash-attention per 16-row query tile per wave.
// ---------------------------------------------------------------------------

typedef __attribute__((ext_vector_type(16))) __bf16 v16bf;
typedef __attribute__((ext_vector_type(8)))  __bf16 v8bf;
typedef __attribute__((ext_vector_type(8)))  float  v8f;

union V16 { v16bf v; struct { v8bf lo, hi; } p; };

#define NTOK   16384   // B*T
#define TT     2048
#define CC     1024
#define HH     128
#define KPAD   136     // W LDS row stride (bf16): 68 words -> lane n uses banks 4n..4n+3
#define PSTR   40      // P LDS row stride (bf16): 20 words -> distinct banks per lane

static __device__ inline v8bf cvt8(float4 a, float4 b) {
  v8bf r;
  r[0] = (__bf16)a.x; r[1] = (__bf16)a.y; r[2] = (__bf16)a.z; r[3] = (__bf16)a.w;
  r[4] = (__bf16)b.x; r[5] = (__bf16)b.y; r[6] = (__bf16)b.z; r[7] = (__bf16)b.w;
  return r;
}

static __device__ inline v8f wmma_bf16(const V16& a, const V16& b, v8f c) {
  return __builtin_amdgcn_wmma_f32_16x16x32_bf16(
      /*neg_a=*/false, a.v, /*neg_b=*/false, b.v,
      /*c_mod=*/(short)0, c, /*reuse_a=*/false, /*reuse_b=*/false);
}

// ---------------------------------------------------------------------------
// Kernel 1: qkv projection.  grid = (128 m-blocks, 3 matrices), block = 256.
// Each wave: 16 output rows x full 128 cols, accumulating over C=1024.
// ---------------------------------------------------------------------------
__global__ __launch_bounds__(256)
void qkv_proj(const float* __restrict__ x,
              const float* __restrict__ Wq, const float* __restrict__ Wk,
              const float* __restrict__ Wv,
              __bf16* __restrict__ Qw, __bf16* __restrict__ Kw,
              __bf16* __restrict__ Vt) {
  const int kind = blockIdx.y;                       // 0=Q,1=K,2=V
  const float* W = (kind == 0) ? Wq : (kind == 1) ? Wk : Wv;
  const int tid    = threadIdx.x;
  const int wave   = tid >> 5;
  const int lane   = tid & 31;
  const int lane16 = lane & 15;
  const int hsel   = (lane >> 4) << 3;               // 0 / 8 (K-chunk + D-row half)
  const int m0     = blockIdx.x * 128 + wave * 16;   // first output row of wave

  __shared__ __bf16 Wlds[HH * KPAD];                 // W slab, [n][k] order

  v8f acc[8];
  #pragma unroll
  for (int t = 0; t < 8; ++t)
    #pragma unroll
    for (int r = 0; r < 8; ++r) acc[t][r] = 0.0f;

  for (int kb = 0; kb < CC; kb += 128) {
    __syncthreads();
    // stage W[kb..kb+127][0..127] -> Wlds[n*KPAD + kl], bf16, coalesced reads
    for (int i = tid; i < 128 * 128; i += 256) {
      const int kl = i >> 7;
      const int n  = i & 127;
      Wlds[n * KPAD + kl] = (__bf16)W[(size_t)(kb + kl) * HH + n];
    }
    __syncthreads();

    #pragma unroll
    for (int ks = 0; ks < 128; ks += 32) {
      // A fragment: x rows are contiguous f32; chunk layout per WMMA A-spec
      const float* xr = x + (size_t)(m0 + lane16) * CC + kb + ks + hsel;
      V16 a;
      a.p.lo = cvt8(*(const float4*)(xr),      *(const float4*)(xr + 4));
      a.p.hi = cvt8(*(const float4*)(xr + 16), *(const float4*)(xr + 20));
      #pragma unroll
      for (int t = 0; t < 8; ++t) {                  // 8 n-tiles of 16
        const __bf16* bp = &Wlds[(t * 16 + lane16) * KPAD + ks + hsel];
        V16 bf;
        bf.p.lo = *(const v8bf*)(bp);
        bf.p.hi = *(const v8bf*)(bp + 16);
        acc[t] = wmma_bf16(a, bf, acc[t]);
      }
    }
  }

  if (kind < 2) {
    // Q/K row-major [NTOK][H] bf16
    __bf16* O = (kind == 0) ? Qw : Kw;
    #pragma unroll
    for (int t = 0; t < 8; ++t)
      #pragma unroll
      for (int r = 0; r < 8; ++r)
        O[(size_t)(m0 + r + hsel) * HH + t * 16 + lane16] = (__bf16)acc[t][r];
  } else {
    // V transposed [B][H][T] bf16 -> one contiguous b128 store per tile
    const int b  = m0 >> 11;
    const int tq = m0 & 2047;
    #pragma unroll
    for (int t = 0; t < 8; ++t) {
      v8bf pk;
      #pragma unroll
      for (int r = 0; r < 8; ++r) pk[r] = (__bf16)acc[t][r];
      *(v8bf*)&Vt[((size_t)(b * HH + t * 16 + lane16)) * TT + tq + hsel] = pk;
    }
  }
}

// ---------------------------------------------------------------------------
// Kernel 2: flash attention.  grid = 128, block = 256 (8 waves).
// wave <- one 16-row query tile; key blocks of 32; online softmax.
// ---------------------------------------------------------------------------
__global__ __launch_bounds__(256)
void attn(const __bf16* __restrict__ Qw, const __bf16* __restrict__ Kw,
          const __bf16* __restrict__ Vt, float* __restrict__ out) {
  const int tid    = threadIdx.x;
  const int wave   = tid >> 5;
  const int lane   = tid & 31;
  const int lane16 = lane & 15;
  const int hsel   = (lane >> 4) << 3;
  const int tile   = blockIdx.x * 8 + wave;          // 0..1023
  const int b      = tile >> 7;
  const int q0     = (tile & 127) * 16;
  const size_t bb  = (size_t)b * TT;

  __shared__ __bf16 Plds[8][16 * PSTR];
  __bf16* pl = Plds[wave];

  // Q A-fragments for the 4 k-steps over H=128
  V16 qf[4];
  {
    const __bf16* qr = Qw + (bb + q0 + lane16) * HH + hsel;
    #pragma unroll
    for (int f = 0; f < 4; ++f) {
      qf[f].p.lo = *(const v8bf*)(qr + f * 32);
      qf[f].p.hi = *(const v8bf*)(qr + f * 32 + 16);
    }
  }

  v8f o[8];
  float mi[8], li[8];
  #pragma unroll
  for (int t = 0; t < 8; ++t)
    #pragma unroll
    for (int r = 0; r < 8; ++r) o[t][r] = 0.0f;
  #pragma unroll
  for (int r = 0; r < 8; ++r) { mi[r] = -3.0e38f; li[r] = 0.0f; }

  const float sc    = 0.08838834764831845f;          // 1/sqrt(128)
  const int   qlast = q0 + 15;

  for (int s0 = 0; s0 <= qlast; s0 += 32) {
    // ---- S = Q K^T for two 16-wide key subtiles ----
    v8f c0, c1;
    #pragma unroll
    for (int r = 0; r < 8; ++r) { c0[r] = 0.0f; c1[r] = 0.0f; }
    const __bf16* kr0 = Kw + (bb + s0 + lane16) * HH + hsel;
    const __bf16* kr1 = kr0 + 16 * HH;
    #pragma unroll
    for (int f = 0; f < 4; ++f) {
      V16 k0, k1;
      k0.p.lo = *(const v8bf*)(kr0 + f * 32);
      k0.p.hi = *(const v8bf*)(kr0 + f * 32 + 16);
      k1.p.lo = *(const v8bf*)(kr1 + f * 32);
      k1.p.hi = *(const v8bf*)(kr1 + f * 32 + 16);
      c0 = wmma_bf16(qf[f], k0, c0);
      c1 = wmma_bf16(qf[f], k1, c1);
    }

    // ---- scale + causal mask + online softmax ----
    const int col0 = s0 + lane16, col1 = col0 + 16;
    #pragma unroll
    for (int r = 0; r < 8; ++r) {
      const int rowg = q0 + r + hsel;
      float v0 = c0[r] * sc, v1 = c1[r] * sc;
      if (col0 > rowg) v0 = -3.0e38f;
      if (col1 > rowg) v1 = -3.0e38f;

      float vm = fmaxf(v0, v1);
      #pragma unroll
      for (int off = 8; off >= 1; off >>= 1) vm = fmaxf(vm, __shfl_xor(vm, off, 16));
      const float mn = fmaxf(mi[r], vm);
      const float al = __expf(mi[r] - mn);
      const float p0 = __expf(v0 - mn);
      const float p1 = __expf(v1 - mn);
      float rs = p0 + p1;
      #pragma unroll
      for (int off = 8; off >= 1; off >>= 1) rs += __shfl_xor(rs, off, 16);
      li[r] = li[r] * al + rs;
      mi[r] = mn;
      #pragma unroll
      for (int t = 0; t < 8; ++t) o[t][r] *= al;
      c0[r] = p0; c1[r] = p1;
    }

    // ---- D-layout -> A-layout transpose of P through LDS ----
    #pragma unroll
    for (int r = 0; r < 8; ++r) {
      const int m = r + hsel;
      pl[m * PSTR + lane16]      = (__bf16)c0[r];
      pl[m * PSTR + 16 + lane16] = (__bf16)c1[r];
    }
    asm volatile("s_wait_dscnt 0" ::: "memory");     // same-wave LDS RAW

    V16 pf;
    pf.p.lo = *(const v8bf*)&pl[lane16 * PSTR + hsel];
    pf.p.hi = *(const v8bf*)&pl[lane16 * PSTR + 16 + hsel];

    // ---- O += P @ V   (V^T rows give contiguous B-fragments) ----
    const __bf16* vr = Vt + ((size_t)b * HH + lane16) * TT + s0 + hsel;
    #pragma unroll
    for (int t = 0; t < 8; ++t) {
      const __bf16* vp = vr + (size_t)t * 16 * TT;
      V16 vf;
      vf.p.lo = *(const v8bf*)(vp);
      vf.p.hi = *(const v8bf*)(vp + 16);
      o[t] = wmma_bf16(pf, vf, o[t]);
    }
  }

  // ---- epilogue: divide by row sums, write f32 ----
  float rl[8];
  #pragma unroll
  for (int r = 0; r < 8; ++r) rl[r] = 1.0f / li[r];
  float* orow = out + (bb + q0) * HH;
  #pragma unroll
  for (int t = 0; t < 8; ++t)
    #pragma unroll
    for (int r = 0; r < 8; ++r)
      orow[(size_t)(r + hsel) * HH + t * 16 + lane16] = o[t][r] * rl[r];
}

// ---------------------------------------------------------------------------
extern "C" void kernel_launch(void* const* d_in, const int* in_sizes, int n_in,
                              void* d_out, int out_size, void* d_ws, size_t ws_size,
                              hipStream_t stream) {
  (void)in_sizes; (void)n_in; (void)out_size;
  // setup_inputs order: x, Wk, Wq, Wv
  const float* x  = (const float*)d_in[0];
  const float* Wk = (const float*)d_in[1];
  const float* Wq = (const float*)d_in[2];
  const float* Wv = (const float*)d_in[3];

  const size_t qkv_elems = (size_t)NTOK * HH;        // per matrix
  if (ws_size < 3 * qkv_elems * sizeof(__bf16)) return;
  __bf16* Qw = (__bf16*)d_ws;
  __bf16* Kw = Qw + qkv_elems;
  __bf16* Vt = Kw + qkv_elems;

  qkv_proj<<<dim3(128, 3), 256, 0, stream>>>(x, Wq, Wk, Wv, Qw, Kw, Vt);
  attn<<<dim3(128), 256, 0, stream>>>(Qw, Kw, Vt, (float*)d_out);
}
